// NonLocalBlock2D_21139829031158
// MI455X (gfx1250) — matrix operand order repaired and checked by reading, not compiled, not run on hardware
//
#include <hip/hip_runtime.h>
#include <hip/hip_bf16.h>

// ---------------------------------------------------------------------------
// NonLocalBlock2D fused implementation for gfx1250 (MI455X).
// B=8, H=W=64 -> N=4096 tokens, C=128, C'=64.
// Pass 0: weight transpose->bf16 + BN constant folding (one-time, tiny).
// Pass 1: bf16 projections theta/phi/g via v_wmma_f32_16x16x32_bf16,
//         outputs written as b128 vector stores (channel-major tiles).
//         theta (incl. bias) is pre-scaled by log2(e) so the attention
//         softmax needs only a bare v_exp_f32 (exp2) per score.
// Pass 2: flash-attention style fused S^T = phi*theta^T, shift-free softmax
//         (softmax is shift-invariant; scores bounded), y += P*g, then
//         W-proj + folded-BN + residual epilogue, all WMMA bf16.
//         Double-buffered LDS; phi staged via global_load_async_to_lds_b128
//         overlapped with WMMA compute; one barrier per key tile.
// ---------------------------------------------------------------------------

typedef __attribute__((ext_vector_type(16))) __bf16 v16bf;
typedef __attribute__((ext_vector_type(8)))  float  v8f;

#define BATCH 8
#define NTOK  4096
#define NC    128
#define INTER 64
#define BN_EPS 1e-4f
#define LOG2E 1.4426950408889634f

union BF16Frag {
    v16bf    v;
    unsigned u[8];
    __bf16   h[16];
    uint4    q[2];
};

static __device__ __forceinline__ v8f wmma_bf16(const v16bf a, const v16bf b, v8f c) {
    return __builtin_amdgcn_wmma_f32_16x16x32_bf16(false, a, false, b, (short)0, c,
                                                   false, false);
}

static __device__ __forceinline__ unsigned pack_bf16(float a, float b) {
    union { __bf16 h[2]; unsigned u; } t;
    t.h[0] = (__bf16)a;
    t.h[1] = (__bf16)b;
    return t.u;
}

// Async copy 16B/lane from global to LDS (gfx1250 ASYNCcnt path).
static __device__ __forceinline__ void async_load_b128(const void* gaddr, void* lds_dst) {
    unsigned lds_off = (unsigned)(unsigned long long)lds_dst;
    asm volatile("global_load_async_to_lds_b128 %0, %1, off"
                 :: "v"(lds_off), "v"(gaddr)
                 : "memory");
}

static __device__ __forceinline__ void wait_asynccnt0() {
    asm volatile("s_wait_asynccnt 0x0" ::: "memory");
}

// ---------------------------------------------------------------------------
// Pass 0: weight prep.
//   twT/pwT/gwT [64][128] bf16 : transposed projection weights (k contiguous).
//   WwT        [128][64] bf16 : transposed output weights (c contiguous).
//   bnA/bnB    [128]     f32  : folded BN:  out = wy*bnA + bnB + x.
// ---------------------------------------------------------------------------
__global__ void __launch_bounds__(256) nlb_prep_kernel(
    const float* __restrict__ tw, const float* __restrict__ pw,
    const float* __restrict__ gw, const float* __restrict__ Ww,
    const float* __restrict__ Wb,
    const float* __restrict__ bn_gamma, const float* __restrict__ bn_beta,
    const float* __restrict__ bn_mean,  const float* __restrict__ bn_var,
    __bf16* __restrict__ twT, __bf16* __restrict__ pwT,
    __bf16* __restrict__ gwT, __bf16* __restrict__ WwT,
    float* __restrict__ bnA, float* __restrict__ bnB)
{
    const int t = blockIdx.x * 256 + threadIdx.x;
    if (t < 64 * 128) {
        const int r = t >> 7, c = t & 127;      // r: inter idx, c: nc idx
        twT[r * 128 + c] = (__bf16)tw[c * 64 + r];
        pwT[r * 128 + c] = (__bf16)pw[c * 64 + r];
        gwT[r * 128 + c] = (__bf16)gw[c * 64 + r];
        const int oc = t >> 6, cc = t & 63;     // oc: nc idx, cc: inter idx
        WwT[oc * 64 + cc] = (__bf16)Ww[cc * 128 + oc];
    }
    if (t < NC) {
        const float inv = bn_gamma[t] * rsqrtf(bn_var[t] + BN_EPS);
        bnA[t] = inv;
        bnB[t] = (Wb[t] - bn_mean[t]) * inv + bn_beta[t];
    }
}

// ---------------------------------------------------------------------------
// Pass 1: out[tok, 0:64] = bf16( (x[tok, :] @ w + bias) * oscale ), computed
// as out^T tiles [16 ch x 16 tok] so stores are contiguous b128 per lane.
// oscale = log2(e) for theta (folds the exp->exp2 conversion), 1.0 otherwise.
// One wave per 16-token block; blockDim 128 (4 waves); grid 512.
// ---------------------------------------------------------------------------
__global__ void __launch_bounds__(128) nlb_proj_kernel(
    const float* __restrict__ x, const __bf16* __restrict__ wT,
    const float* __restrict__ bias, float oscale, __bf16* __restrict__ out)
{
    const int wave = threadIdx.x >> 5;
    const int lane = threadIdx.x & 31;
    const int n    = lane & 15;                 // token col / ch row selector
    const int hi   = lane >> 4;
    const int tok0 = (blockIdx.x * 4 + wave) * 16;

    v8f acc[4];
#pragma unroll
    for (int ct = 0; ct < 4; ++ct)
#pragma unroll
        for (int v = 0; v < 8; ++v) acc[ct][v] = 0.0f;

#pragma unroll
    for (int kk = 0; kk < 4; ++kk) {            // K = 128 in chunks of 32
        // B fragment: x row (token n), 16 consecutive fp32 -> bf16.
        BF16Frag bx;
        const float* xa = x + (size_t)(tok0 + n) * NC + kk * 32 + hi * 16;
#pragma unroll
        for (int i = 0; i < 16; ++i) bx.h[i] = (__bf16)xa[i];

#pragma unroll
        for (int ct = 0; ct < 4; ++ct) {        // 4 x 16 output channels (M)
            // A fragment: wT row (channel ct*16+n), contiguous b128 loads.
            const uint4* wp = (const uint4*)(wT + (size_t)(ct * 16 + n) * 128 + kk * 32);
            BF16Frag aw;
            aw.q[0] = wp[2 * hi];
            aw.q[1] = wp[2 * hi + 1];
            acc[ct] = wmma_bf16(aw.v, bx.v, acc[ct]);
        }
    }

    // C/D: lane column = token n, rows = channels (hi*8 + v).
#pragma unroll
    for (int ct = 0; ct < 4; ++ct) {
        const int cbase = ct * 16 + hi * 8;
        const float4* bp = (const float4*)(bias + cbase);
        const float4 b0 = bp[0], b1 = bp[1];
        uint4 st;
        st.x = pack_bf16((acc[ct][0] + b0.x) * oscale, (acc[ct][1] + b0.y) * oscale);
        st.y = pack_bf16((acc[ct][2] + b0.z) * oscale, (acc[ct][3] + b0.w) * oscale);
        st.z = pack_bf16((acc[ct][4] + b1.x) * oscale, (acc[ct][5] + b1.y) * oscale);
        st.w = pack_bf16((acc[ct][6] + b1.z) * oscale, (acc[ct][7] + b1.w) * oscale);
        *(uint4*)(out + (size_t)(tok0 + n) * INTER + cbase) = st;
    }
}

// ---------------------------------------------------------------------------
// Pass 2: fused attention + output projection + folded BN + residual.
// Each wave owns 16 queries; workgroup (4 waves) shares double-buffered LDS:
// per buffer, phi tile [32 keys x 64] (async DMA) + g^T tile [64 x 32 keys].
// theta is pre-scaled by log2(e) -> P = exp2(S') with a single v_exp_f32.
// grid = (4096/64, 8), block = 128.
// ---------------------------------------------------------------------------
__global__ void __launch_bounds__(128) nlb_attn_kernel(
    const __bf16* __restrict__ theta, const __bf16* __restrict__ phi,
    const __bf16* __restrict__ g,
    const float* __restrict__ x,
    const __bf16* __restrict__ WwT,
    const float* __restrict__ bnA, const float* __restrict__ bnB,
    float* __restrict__ out)
{
    __shared__ uint4 smem4[1024];                      // 16 KB, two 8 KB buffers
    __bf16* sbase = (__bf16*)smem4;

    const int lane = threadIdx.x & 31;
    const int n    = lane & 15;                        // query / N column
    const int hi   = lane >> 4;
    const int b    = blockIdx.y;
    const int q0   = (blockIdx.x * 4 + (threadIdx.x >> 5)) * 16;
    const size_t base = (size_t)b * NTOK * INTER;

    const int keyloc = threadIdx.x >> 2;               // 0..31 key in tile
    const int part   = (threadIdx.x & 3) * 16;         // 16-ch slice

    // theta as B-fragments [K x 16 queries]; head dim 64 -> two 32-chunks.
    BF16Frag tb0, tb1;
    {
        const uint4* tp0 = (const uint4*)(theta + base + (size_t)(q0 + n) * INTER + hi * 16);
        const uint4* tp1 = (const uint4*)(theta + base + (size_t)(q0 + n) * INTER + 32 + hi * 16);
        tb0.q[0] = tp0[0]; tb0.q[1] = tp0[1];
        tb1.q[0] = tp1[0]; tb1.q[1] = tp1[1];
    }

    float lsum = 0.0f;
    v8f yacc[4];
#pragma unroll
    for (int ct = 0; ct < 4; ++ct)
#pragma unroll
        for (int v = 0; v < 8; ++v) yacc[ct][v] = 0.0f;

    // ---- prologue: stage key tile 0 into buffer 0.
    {
        const __bf16* ps = phi + base + (size_t)keyloc * INTER + part;
        async_load_b128(ps, sbase + keyloc * 64 + part);
        async_load_b128(ps + 8, sbase + keyloc * 64 + part + 8);
        const uint4* gs = (const uint4*)(g + base + (size_t)keyloc * INTER + part);
        BF16Frag gf;
        gf.q[0] = gs[0]; gf.q[1] = gs[1];
        __bf16* sgt0 = sbase + 2048;
#pragma unroll
        for (int i = 0; i < 16; ++i)
            sgt0[(part + i) * 32 + keyloc] = gf.h[i];
        wait_asynccnt0();
    }
    __syncthreads();

    for (int kt = 0; kt < NTOK / 32; ++kt) {
        __bf16* sphi = sbase + (kt & 1) * 4096;
        __bf16* sgt  = sphi + 2048;
        __bf16* nphi = sbase + ((kt + 1) & 1) * 4096;
        __bf16* ngt  = nphi + 2048;
        const bool more = (kt + 1) < (NTOK / 32);

        // ---- kick off next tile's staging (overlaps with compute below).
        BF16Frag gnext;
        if (more) {
            const size_t nb = base + (size_t)((kt + 1) * 32 + keyloc) * INTER + part;
            async_load_b128(phi + nb, nphi + keyloc * 64 + part);
            async_load_b128(phi + nb + 8, nphi + keyloc * 64 + part + 8);
            const uint4* gs = (const uint4*)(g + nb);
            gnext.q[0] = gs[0]; gnext.q[1] = gs[1];
        }

        // ---- S^T tiles: [16 keys x 16 queries], keys along register dim.
        v8f s[2];
#pragma unroll
        for (int st = 0; st < 2; ++st) {
            const uint4* pa = (const uint4*)(sphi + (st * 16 + n) * 64);
            BF16Frag a0, a1;
            a0.q[0] = pa[2 * hi];     a0.q[1] = pa[2 * hi + 1];   // k = 0..31
            a1.q[0] = pa[4 + 2 * hi]; a1.q[1] = pa[5 + 2 * hi];   // k = 32..63
            v8f z;
#pragma unroll
            for (int v = 0; v < 8; ++v) z[v] = 0.0f;
            z = wmma_bf16(a1.v, tb1.v, z);
            z = wmma_bf16(a0.v, tb0.v, z);
            s[st] = z;
        }

        // ---- shift-free softmax numerator: P = exp2(S') where S' already
        // carries the log2(e) factor (folded into theta).  Softmax is
        // shift-invariant and |S| is bounded well inside fp32 exp range.
        float rs = 0.0f;
        unsigned q0d[4], q1d[4];
#pragma unroll
        for (int j = 0; j < 4; ++j) {
            const float p0a = __builtin_amdgcn_exp2f(s[0][2 * j]);
            const float p0b = __builtin_amdgcn_exp2f(s[0][2 * j + 1]);
            const float p1a = __builtin_amdgcn_exp2f(s[1][2 * j]);
            const float p1b = __builtin_amdgcn_exp2f(s[1][2 * j + 1]);
            rs += p0a + p0b + p1a + p1b;
            q0d[j] = pack_bf16(p0a, p0b);
            q1d[j] = pack_bf16(p1a, p1b);
        }
        lsum += rs;     // per-lane partial; lane-pair combine once at the end

        // Re-pack P^T (C/D layout) into a B-fragment [32 keys x 16 queries].
        BF16Frag pb;
#pragma unroll
        for (int j = 0; j < 4; ++j) {
            const unsigned q0p = (unsigned)__shfl_xor((int)q0d[j], 16, 32);
            const unsigned q1p = (unsigned)__shfl_xor((int)q1d[j], 16, 32);
            pb.u[j]     = hi ? q1p    : q0d[j];
            pb.u[4 + j] = hi ? q1d[j] : q0p;
        }

        // ---- y^T[c, q] += g^T[c, keys] * P^T[keys, q]
#pragma unroll
        for (int ct = 0; ct < 4; ++ct) {
            const uint4* gp = (const uint4*)(sgt + (ct * 16 + n) * 32);
            BF16Frag ga;
            ga.q[0] = gp[2 * hi];
            ga.q[1] = gp[2 * hi + 1];
            yacc[ct] = wmma_bf16(ga.v, pb.v, yacc[ct]);
        }

        // ---- finish staging of next tile, single barrier per iteration.
        if (more) {
#pragma unroll
            for (int i = 0; i < 16; ++i)
                ngt[(part + i) * 32 + keyloc] = gnext.h[i];
        }
        wait_asynccnt0();
        __syncthreads();
    }

    // ---- normalize (lane pair holds same query -> combine partial sums once)
    lsum += __shfl_xor(lsum, 16, 32);
    const float linv = 1.0f / lsum;
#pragma unroll
    for (int ct = 0; ct < 4; ++ct)
#pragma unroll
        for (int v = 0; v < 8; ++v) yacc[ct][v] *= linv;

    // ---- convert y^T accumulators into B-fragments (c along K).
    BF16Frag by0, by1;
#pragma unroll
    for (int j = 0; j < 4; ++j) {
        const unsigned d0 = pack_bf16(yacc[0][2 * j], yacc[0][2 * j + 1]);
        const unsigned d1 = pack_bf16(yacc[1][2 * j], yacc[1][2 * j + 1]);
        const unsigned d2 = pack_bf16(yacc[2][2 * j], yacc[2][2 * j + 1]);
        const unsigned d3 = pack_bf16(yacc[3][2 * j], yacc[3][2 * j + 1]);
        const unsigned d0p = (unsigned)__shfl_xor((int)d0, 16, 32);
        const unsigned d1p = (unsigned)__shfl_xor((int)d1, 16, 32);
        const unsigned d2p = (unsigned)__shfl_xor((int)d2, 16, 32);
        const unsigned d3p = (unsigned)__shfl_xor((int)d3, 16, 32);
        by0.u[j]     = hi ? d1p : d0;
        by0.u[4 + j] = hi ? d1  : d0p;
        by1.u[j]     = hi ? d3p : d2;
        by1.u[4 + j] = hi ? d3  : d2p;
    }

    // ---- epilogue: wy^T[oc, q] = W^T[oc, c] * y^T[c, q]; out = wy*A + B + x.
    const size_t xb = (size_t)b * NTOK * NC;
#pragma unroll
    for (int ot = 0; ot < 8; ++ot) {
        const int oc = ot * 16 + n;
        const uint4* wp = (const uint4*)(WwT + (size_t)oc * 64);
        BF16Frag aw0, aw1;
        aw0.q[0] = wp[2 * hi];     aw0.q[1] = wp[2 * hi + 1];   // c = 0..31
        aw1.q[0] = wp[4 + 2 * hi]; aw1.q[1] = wp[5 + 2 * hi];   // c = 32..63

        v8f wy;
#pragma unroll
        for (int v = 0; v < 8; ++v) wy[v] = 0.0f;
        wy = wmma_bf16(aw1.v, by1.v, wy);
        wy = wmma_bf16(aw0.v, by0.v, wy);

        // C/D rows = 8 consecutive output channels -> float4 epilogue.
        const int cbase = ot * 16 + hi * 8;
        const float4 a0 = *(const float4*)(bnA + cbase);
        const float4 a1 = *(const float4*)(bnA + cbase + 4);
        const float4 s0 = *(const float4*)(bnB + cbase);
        const float4 s1 = *(const float4*)(bnB + cbase + 4);
        const size_t idx = xb + (size_t)(q0 + n) * NC + cbase;
        const float4 x0 = *(const float4*)(x + idx);
        const float4 x1 = *(const float4*)(x + idx + 4);
        float4 o0, o1;
        o0.x = wy[0] * a0.x + s0.x + x0.x;
        o0.y = wy[1] * a0.y + s0.y + x0.y;
        o0.z = wy[2] * a0.z + s0.z + x0.z;
        o0.w = wy[3] * a0.w + s0.w + x0.w;
        o1.x = wy[4] * a1.x + s1.x + x1.x;
        o1.y = wy[5] * a1.y + s1.y + x1.y;
        o1.z = wy[6] * a1.z + s1.z + x1.z;
        o1.w = wy[7] * a1.w + s1.w + x1.w;
        *(float4*)(out + idx)     = o0;
        *(float4*)(out + idx + 4) = o1;
    }
}

// ---------------------------------------------------------------------------
extern "C" void kernel_launch(void* const* d_in, const int* in_sizes, int n_in,
                              void* d_out, int out_size, void* d_ws, size_t ws_size,
                              hipStream_t stream)
{
    (void)in_sizes; (void)n_in; (void)out_size; (void)ws_size;

    const float* x        = (const float*)d_in[0];
    const float* theta_w  = (const float*)d_in[1];
    const float* theta_b  = (const float*)d_in[2];
    const float* phi_w    = (const float*)d_in[3];
    const float* phi_b    = (const float*)d_in[4];
    const float* g_w      = (const float*)d_in[5];
    const float* g_b      = (const float*)d_in[6];
    const float* W_w      = (const float*)d_in[7];
    const float* W_b      = (const float*)d_in[8];
    const float* bn_gamma = (const float*)d_in[9];
    const float* bn_beta  = (const float*)d_in[10];
    const float* bn_mean  = (const float*)d_in[11];
    const float* bn_var   = (const float*)d_in[12];
    float* out = (float*)d_out;

    const size_t proj_elems = (size_t)BATCH * NTOK * INTER;  // 2M elems each
    __bf16* theta = (__bf16*)d_ws;
    __bf16* phi   = theta + proj_elems;
    __bf16* g     = phi + proj_elems;
    __bf16* twT   = g + proj_elems;          // [64][128]
    __bf16* pwT   = twT + 64 * 128;
    __bf16* gwT   = pwT + 64 * 128;
    __bf16* WwT   = gwT + 64 * 128;          // [128][64]
    float*  bnA   = (float*)(WwT + 128 * 64);
    float*  bnB   = bnA + NC;

    // Pass 0: weight transpose + bf16 convert + BN folding.
    hipLaunchKernelGGL(nlb_prep_kernel, dim3(32), dim3(256), 0, stream,
                       theta_w, phi_w, g_w, W_w, W_b,
                       bn_gamma, bn_beta, bn_mean, bn_var,
                       twT, pwT, gwT, WwT, bnA, bnB);

    // Pass 1: projections (32768 tokens / (4 waves * 16) = 512 blocks).
    // theta carries the log2(e) softmax fold.
    dim3 pgrid(512), pblk(128);
    hipLaunchKernelGGL(nlb_proj_kernel, pgrid, pblk, 0, stream, x, twT, theta_b, LOG2E, theta);
    hipLaunchKernelGGL(nlb_proj_kernel, pgrid, pblk, 0, stream, x, pwT, phi_b,   1.0f,  phi);
    hipLaunchKernelGGL(nlb_proj_kernel, pgrid, pblk, 0, stream, x, gwT, g_b,     1.0f,  g);

    // Pass 2: fused attention + epilogue.
    dim3 agrid(NTOK / 64, BATCH), ablk(128);
    hipLaunchKernelGGL(nlb_attn_kernel, agrid, ablk, 0, stream,
                       theta, phi, g, x, WwT, bnA, bnB, out);
}
